// ImageTransformerLayer_64218351010457
// MI455X (gfx1250) — matrix operand order repaired and checked by reading, not compiled, not run on hardware
//
#include <hip/hip_runtime.h>

typedef __attribute__((ext_vector_type(16))) __bf16 v16bf;
typedef __attribute__((ext_vector_type(8)))  __bf16 v8bf;
typedef __attribute__((ext_vector_type(8)))  float  v8f;

// ---------------------------------------------------------------------------
// elementwise fp32 -> bf16
// ---------------------------------------------------------------------------
__global__ void cvt_bf16_kernel(const float* __restrict__ in,
                                __bf16* __restrict__ out, long long n) {
  long long i = (long long)blockIdx.x * blockDim.x + threadIdx.x;
  long long stride = (long long)gridDim.x * blockDim.x;
  for (; i < n; i += stride) out[i] = (__bf16)in[i];
}

// tiled transpose + fp32 -> bf16 : out[n][k] = in[k][n]  (in is [K][N])
__global__ __launch_bounds__(256)
void cvt_transpose_kernel(const float* __restrict__ in, __bf16* __restrict__ out,
                          int K, int N) {
  __shared__ __bf16 tile[32][33];
  int nb = blockIdx.x * 32, kb = blockIdx.y * 32;
  int tx = threadIdx.x, ty = threadIdx.y;          // 32 x 8
#pragma unroll
  for (int j = 0; j < 32; j += 8)
    tile[ty + j][tx] = (__bf16)in[(long long)(kb + ty + j) * N + (nb + tx)];
  __syncthreads();
#pragma unroll
  for (int j = 0; j < 32; j += 8)
    out[(long long)(nb + ty + j) * K + (kb + tx)] = tile[tx][ty + j];
}

// KP[b,p,k] = bf16(Kf[b,p,k] + P[p,k])   (P shared across batch)
__global__ void kp_kernel(const float* __restrict__ Kf, const float* __restrict__ P,
                          __bf16* __restrict__ KP, long long n, long long per_batch) {
  long long i = (long long)blockIdx.x * blockDim.x + threadIdx.x;
  long long stride = (long long)gridDim.x * blockDim.x;
  for (; i < n; i += stride) {
    long long rem = i % per_batch;
    KP[i] = (__bf16)(Kf[i] + P[rem]);
  }
}

// ---------------------------------------------------------------------------
// block reductions (256 threads)
// ---------------------------------------------------------------------------
static __device__ __forceinline__ float block_reduce_sum(float v, float* red) {
  int tid = threadIdx.x;
  __syncthreads();
  red[tid] = v;
  __syncthreads();
  for (int s = 128; s > 0; s >>= 1) {
    if (tid < s) red[tid] += red[tid + s];
    __syncthreads();
  }
  return red[0];
}
static __device__ __forceinline__ float block_reduce_max(float v, float* red) {
  int tid = threadIdx.x;
  __syncthreads();
  red[tid] = v;
  __syncthreads();
  for (int s = 128; s > 0; s >>= 1) {
    if (tid < s) red[tid] = fmaxf(red[tid], red[tid + s]);
    __syncthreads();
  }
  return red[0];
}

// ---------------------------------------------------------------------------
// Tiled bf16 WMMA GEMM: C[M,N] = epi(scale * A[M,K] @ B(+T) + bias) (+resid)
// Block tile 128x128x32, 256 threads = 8 waves (4x2), wave tile 32x64
// (2x4 = 8 v_wmma_f32_16x16x32_bf16 per wave per k-step).
// Double-buffered LDS (compile-time buffer indices, 1 barrier per k-step);
// staging is a fused global->reg->LDS step with short live ranges (no spills).
// Cross-wave co-execution (2 waves/SIMD) hides staging latency under WMMA.
// Requires K % 64 == 0 (true for all uses).
// ---------------------------------------------------------------------------
template <bool TRANSB, bool OUT_BF16, bool RELU>
__global__ __launch_bounds__(256)
void gemm_bf16_wmma(const __bf16* __restrict__ A, const __bf16* __restrict__ B,
                    const float* __restrict__ bias, const float* __restrict__ resid,
                    float* __restrict__ Cf, __bf16* __restrict__ Cb,
                    int N, int K, float scale,
                    long long strideA, long long strideB, long long strideC) {
  constexpr int BM = 128, BN = 128, BK = 32;
  __shared__ __attribute__((aligned(16))) __bf16 As[2][BM][BK + 8];  // [m][k]
  __shared__ __attribute__((aligned(16))) __bf16 Bs[2][BN][BK + 8];  // [n][k]

  const long long bz = blockIdx.z;
  const __bf16* Ab = A + bz * strideA;
  const __bf16* Bb = B + bz * strideB;
  const long long coff = bz * strideC;

  const int tid  = threadIdx.x;
  const int wave = tid >> 5;
  const int lane = tid & 31;
  const int wm = wave >> 1, wn = wave & 1;    // 4 x 2 wave grid
  const int half = lane >> 4, l16 = lane & 15;
  const int m0 = blockIdx.y * BM;
  const int n0 = blockIdx.x * BN;

  v8f acc[2][4] = {};

  // fused stage: global -> regs -> LDS, short live ranges
  auto stage = [&](int buf, int k0) {
    uint4 ar[2], br[2];
#pragma unroll
    for (int c = 0; c < 2; ++c) {
      int chunk = tid * 2 + c;
      int r  = chunk >> 2;                    // 0..127
      int cc = (chunk & 3) << 3;              // 0,8,16,24
      ar[c] = *(const uint4*)(Ab + (long long)(m0 + r) * K + (k0 + cc));
    }
    if (!TRANSB) {                            // B[K][N]: coalesced along n
#pragma unroll
      for (int c = 0; c < 2; ++c) {
        int chunk = tid * 2 + c;
        int r  = chunk >> 4;                  // k row 0..31
        int cc = (chunk & 15) << 3;           // n chunk
        br[c] = *(const uint4*)(Bb + (long long)(k0 + r) * N + (n0 + cc));
      }
    } else {                                  // B[N][K]: K-contiguous already
#pragma unroll
      for (int c = 0; c < 2; ++c) {
        int chunk = tid * 2 + c;
        int nn = chunk >> 2;                  // 0..127
        int kc = (chunk & 3) << 3;
        br[c] = *(const uint4*)(Bb + (long long)(n0 + nn) * K + (k0 + kc));
      }
    }
#pragma unroll
    for (int c = 0; c < 2; ++c) {
      int chunk = tid * 2 + c;
      int r  = chunk >> 2;
      int cc = (chunk & 3) << 3;
      *(uint4*)(&As[buf][r][cc]) = ar[c];
    }
    if (!TRANSB) {                            // scatter-transpose into Bs[n][k]
#pragma unroll
      for (int c = 0; c < 2; ++c) {
        int chunk = tid * 2 + c;
        int r  = chunk >> 4;
        int cc = (chunk & 15) << 3;
        v8bf t = *(v8bf*)&br[c];
#pragma unroll
        for (int j = 0; j < 8; ++j) Bs[buf][cc + j][r] = t[j];
      }
    } else {                                  // direct vector copy
#pragma unroll
      for (int c = 0; c < 2; ++c) {
        int chunk = tid * 2 + c;
        int nn = chunk >> 2;
        int kc = (chunk & 3) << 3;
        *(uint4*)(&Bs[buf][nn][kc]) = br[c];
      }
    }
  };
  auto compute = [&](int buf) {               // called with literal 0/1 only
    v16bf af[2], bfr[4];
#pragma unroll
    for (int mi = 0; mi < 2; ++mi) {
      const __bf16* ap = &As[buf][wm * 32 + mi * 16 + l16][0];
      *(uint4*)(&af[mi])       = *(const uint4*)(ap + half * 8);
      *((uint4*)(&af[mi]) + 1) = *(const uint4*)(ap + 16 + half * 8);
    }
#pragma unroll
    for (int ni = 0; ni < 4; ++ni) {
      const __bf16* bp = &Bs[buf][wn * 64 + ni * 16 + l16][0];
      *(uint4*)(&bfr[ni])       = *(const uint4*)(bp + half * 8);
      *((uint4*)(&bfr[ni]) + 1) = *(const uint4*)(bp + 16 + half * 8);
    }
#pragma unroll
    for (int mi = 0; mi < 2; ++mi)
#pragma unroll
      for (int ni = 0; ni < 4; ++ni)
        acc[mi][ni] = __builtin_amdgcn_wmma_f32_16x16x32_bf16(
            false, af[mi], false, bfr[ni], (short)0, acc[mi][ni], false, false);
  };

  stage(0, 0);
  __syncthreads();

  const int nt = K / BK;                      // even for all our K
  for (int t = 0; t < nt; t += 2) {
    stage(1, (t + 1) * BK);                   // other waves' WMMA covers stalls
    if (t + 2 < nt)
      __builtin_prefetch(Ab + (long long)(m0 + (tid & 127)) * K + (t + 2) * BK, 0, 0);
    compute(0);
    __syncthreads();

    if (t + 2 < nt) stage(0, (t + 2) * BK);
    compute(1);
    __syncthreads();
  }

  // ---- epilogue: scale, bias, relu, residual ----
#pragma unroll
  for (int mi = 0; mi < 2; ++mi) {
#pragma unroll
    for (int ni = 0; ni < 4; ++ni) {
      int n = n0 + wn * 64 + ni * 16 + l16;
      float bv = bias ? bias[n] : 0.f;
#pragma unroll
      for (int v = 0; v < 8; ++v) {
        int m = m0 + wm * 32 + mi * 16 + half * 8 + v;
        long long idx = coff + (long long)m * N + n;
        float val = acc[mi][ni][v] * scale + bv;
        if (RELU) val = fmaxf(val, 0.f);
        if (resid) val += resid[idx];
        if (OUT_BF16) Cb[idx] = (__bf16)val;
        else          Cf[idx] = val;
      }
    }
  }
}

// ---------------------------------------------------------------------------
// Fused softmax over 4096 keys + sum over key-depth: A2[h] = sum_g alpha[g,h].
// ---------------------------------------------------------------------------
__global__ __launch_bounds__(256)
void softmax_a2_kernel(const float* __restrict__ E, __bf16* __restrict__ A2, int b) {
  __shared__ float red[256];
  int p = blockIdx.x;                       // query row within batch: d*256+f
  const float* e = E + (long long)p * 4096;
  int tid = threadIdx.x;
  float loc[16];
  float lmax = -3.4e38f;
#pragma unroll
  for (int j = 0; j < 16; ++j) {
    loc[j] = e[tid + j * 256];
    lmax = fmaxf(lmax, loc[j]);
  }
  float gmax = block_reduce_max(lmax, red);
  float pes = 0.f;
#pragma unroll
  for (int j = 0; j < 16; ++j) pes += __expf(loc[j] - gmax);
  float tot = block_reduce_sum(pes, red);
  int d = p >> 8, f = p & 255;
  long long idx = (((long long)(b * 16 + d)) * 256 + f) * 256 + tid;
  A2[idx] = (__bf16)(pes / tot);            // == sum_g alpha[g, h=tid]
}

// ---------------------------------------------------------------------------
// LayerNorm over NC=4096 channels; optional bf16 secondary output.
// ---------------------------------------------------------------------------
__global__ __launch_bounds__(256)
void layernorm_kernel(const float* __restrict__ X, const float* __restrict__ gam,
                      const float* __restrict__ bet, float* __restrict__ outF,
                      __bf16* __restrict__ outB) {
  __shared__ float red[256];
  const int NCc = 4096;
  long long row = blockIdx.x;
  const float* x = X + row * NCc;
  int tid = threadIdx.x;
  float loc[16];
  float s = 0.f, ss = 0.f;
#pragma unroll
  for (int j = 0; j < 16; ++j) {
    float v = x[tid + j * 256];
    loc[j] = v; s += v; ss += v * v;
  }
  float mean = block_reduce_sum(s, red) * (1.f / 4096.f);
  float m2   = block_reduce_sum(ss, red) * (1.f / 4096.f);
  float rstd = rsqrtf(m2 - mean * mean + 1e-3f);
#pragma unroll
  for (int j = 0; j < 16; ++j) {
    int c = tid + j * 256;
    float y = (loc[j] - mean) * rstd * gam[c] + bet[c];
    long long idx = row * NCc + c;
    if (outF) outF[idx] = y;
    if (outB) outB[idx] = (__bf16)y;
  }
}

// ---------------------------------------------------------------------------
extern "C" void kernel_launch(void* const* d_in, const int* in_sizes, int n_in,
                              void* d_out, int out_size, void* d_ws, size_t ws_size,
                              hipStream_t stream) {
  (void)in_sizes; (void)n_in; (void)out_size; (void)ws_size;
  const float* X    = (const float*)d_in[0];
  const float* Wq   = (const float*)d_in[1];
  const float* bq   = (const float*)d_in[2];
  const float* Wk   = (const float*)d_in[3];
  const float* bk   = (const float*)d_in[4];
  const float* Wv   = (const float*)d_in[5];
  const float* bv   = (const float*)d_in[6];
  const float* Wmlp = (const float*)d_in[7];
  const float* bmlp = (const float*)d_in[8];
  const float* P    = (const float*)d_in[9];
  const float* g1   = (const float*)d_in[10];
  const float* b1   = (const float*)d_in[11];
  const float* g2   = (const float*)d_in[12];
  const float* b2   = (const float*)d_in[13];
  float* out = (float*)d_out;

  const long long NC = 4096, R = 512, Mrows = 8192;   // Mrows = B*D*F

  char* ws = (char*)d_ws;
  size_t off = 0;
  auto alloc = [&](size_t bytes) {
    void* p = ws + off;
    off += (bytes + 255) & ~(size_t)255;
    return p;
  };
  __bf16* Xb  = (__bf16*)alloc(Mrows * NC * 2);
  __bf16* Wqt = (__bf16*)alloc(NC * R * 2);           // [R][NC]  (transposed)
  __bf16* Wkt = (__bf16*)alloc(NC * R * 2);           // [R][NC]
  __bf16* Wvt = (__bf16*)alloc(NC * NC * 2);          // [NC][NC] (transposed)
  __bf16* Wmt = (__bf16*)alloc(NC * NC * 2);          // [NC][NC]
  __bf16* Qb  = (__bf16*)alloc(Mrows * R * 2);
  float*  Kf  = (float*) alloc(Mrows * R * 4);
  __bf16* KPb = (__bf16*)alloc(Mrows * R * 2);
  float*  E   = (float*) alloc(4096LL * 4096 * 4);    // reused per batch
  __bf16* A2b = (__bf16*)alloc(32LL * 256 * 256 * 2);
  __bf16* Vb  = (__bf16*)alloc(Mrows * NC * 2);
  float*  aX  = (float*) alloc(Mrows * NC * 4);       // attn + X
  float*  y1f = (float*) alloc(Mrows * NC * 4);
  __bf16* y1b = (__bf16*)alloc(Mrows * NC * 2);
  float*  mlr = (float*) alloc(Mrows * NC * 4);       // y1 + relu(mlp)

  dim3 tblk(32, 8);
  cvt_bf16_kernel<<<4096, 256, 0, stream>>>(X, Xb, Mrows * NC);
  cvt_transpose_kernel<<<dim3(16, 128),  tblk, 0, stream>>>(Wq,   Wqt, (int)NC, (int)R);
  cvt_transpose_kernel<<<dim3(16, 128),  tblk, 0, stream>>>(Wk,   Wkt, (int)NC, (int)R);
  cvt_transpose_kernel<<<dim3(128, 128), tblk, 0, stream>>>(Wv,   Wvt, (int)NC, (int)NC);
  cvt_transpose_kernel<<<dim3(128, 128), tblk, 0, stream>>>(Wmlp, Wmt, (int)NC, (int)NC);

  dim3 blk(256);
  // Q = X@Wq + bq  (bf16 out)  [B transposed: direct-copy staging]
  gemm_bf16_wmma<true, true, false><<<dim3(4, 64, 1), blk, 0, stream>>>(
      Xb, Wqt, bq, nullptr, nullptr, Qb, (int)R, (int)NC, 1.f, 0, 0, 0);
  // K = X@Wk + bk  (f32 out, feeds KP)
  gemm_bf16_wmma<true, false, false><<<dim3(4, 64, 1), blk, 0, stream>>>(
      Xb, Wkt, bk, nullptr, Kf, nullptr, (int)R, (int)NC, 1.f, 0, 0, 0);
  // V = X@Wv + bv  (bf16 out)
  gemm_bf16_wmma<true, true, false><<<dim3(32, 64, 1), blk, 0, stream>>>(
      Xb, Wvt, bv, nullptr, nullptr, Vb, (int)NC, (int)NC, 1.f, 0, 0, 0);
  // KP = K + P
  kp_kernel<<<4096, 256, 0, stream>>>(Kf, P, KPb, Mrows * R, 4096LL * R);

  const float esc = 0.044194173824159216f;  // 1/sqrt(512)
  for (int b = 0; b < 2; ++b) {
    // E = scale * Q[b] @ (K[b]+P)^T   (fused scores+pos, NT GEMM)
    gemm_bf16_wmma<true, false, false><<<dim3(32, 32, 1), blk, 0, stream>>>(
        Qb + (long long)b * 4096 * R, KPb + (long long)b * 4096 * R,
        nullptr, nullptr, E, nullptr, 4096, (int)R, esc, 0, 0, 0);
    softmax_a2_kernel<<<4096, 256, 0, stream>>>(E, A2b, b);
  }

  // attnX = A2 @ V + X   (batched over 32 (b,d) slices; NN path)
  gemm_bf16_wmma<false, false, false><<<dim3(32, 2, 32), blk, 0, stream>>>(
      A2b, Vb, nullptr, X, aX, nullptr, (int)NC, 256, 1.f,
      256LL * 256, 256LL * NC, 256LL * NC);

  layernorm_kernel<<<8192, 256, 0, stream>>>(aX, g1, b1, y1f, y1b);

  // mlr = y1 + relu(y1 @ Wmlp + bmlp)
  gemm_bf16_wmma<true, false, true><<<dim3(32, 64, 1), blk, 0, stream>>>(
      y1b, Wmt, bmlp, y1f, mlr, nullptr, (int)NC, (int)NC, 1.f, 0, 0, 0);

  layernorm_kernel<<<8192, 256, 0, stream>>>(mlr, g2, b2, out, nullptr);
}
